// InterestGroupIdentificationModule_22720376995901
// MI455X (gfx1250) — compile-verified, matrix operand order checked by reading, not run on hardware
//
#include <hip/hip_runtime.h>
#include <hip/hip_bf16.h>
#include <math.h>

// Problem constants (from reference)
#define BATCH 256
#define LSEQ  256
#define HDIM  128
#define KCAP  8
#define RITERS 3

typedef __attribute__((ext_vector_type(16))) __bf16 v16bf;
typedef __attribute__((ext_vector_type(8)))  __bf16 v8bf;
typedef __attribute__((ext_vector_type(8)))  float  v8f;

// ---- LDS layout (float offsets), dynamic shared, 320KB total ----
// [0     , 32768) sE    f32 [L][H]                (live whole kernel)
// [32768 , 65536) sHat  f32 [L][H]                (live from GEMM to routing)
// [65536 , 81920) union 64KB:
//    GEMM phase : eBF bf16[L][H] @65536 (64KB), wBF bf16[H][H] @73728 (32KB)
//    post phase : sB@65536(2048) sC@67584(2048) sCap@69632(1024) sMu@70656(1024)
//                 sInv@71680(256) sStr@71936(8) sCons@71944(8) sPair@71952(64)
//                 sPerm@72016(8)   -- all inside the old eBF region
#define OFF_E    0
#define OFF_HAT  32768
#define OFF_EBF  65536
#define OFF_WBF  73728
#define OFF_B    65536
#define OFF_C    67584
#define OFF_CAP  69632
#define OFF_MU   70656
#define OFF_INV  71680
#define OFF_STR  71936
#define OFF_CONS 71944
#define OFF_PAIR 71952
#define OFF_PERM 72016
#define SMEM_FLOATS 81920
#define SMEM_BYTES  (SMEM_FLOATS * 4)   // 327,680 = 320KB (WGP max)

__device__ inline float waveSum(float v) {
#pragma unroll
  for (int off = 16; off > 0; off >>= 1) v += __shfl_xor(v, off, 32);
  return v;
}
__device__ inline float waveMax(float v) {
#pragma unroll
  for (int off = 16; off > 0; off >>= 1) v = fmaxf(v, __shfl_xor(v, off, 32));
  return v;
}

__global__ void __launch_bounds__(256)
mind_fused_kernel(const float* __restrict__ uf,   // [B,L,H]
                  const float* __restrict__ Wp,   // [H,H]
                  const float* __restrict__ binit,// [B,K,L]
                  float* __restrict__ out)        // [B] optimal_k, then [B,7] scores
{
  extern __shared__ float smem[];
  float*  sE   = smem + OFF_E;
  float*  sHat = smem + OFF_HAT;
  __bf16* eBF  = (__bf16*)(smem + OFF_EBF);
  __bf16* wBF  = (__bf16*)(smem + OFF_WBF);
  float*  sB   = smem + OFF_B;
  float*  sC   = smem + OFF_C;
  float*  sCap = smem + OFF_CAP;
  float*  sMu  = smem + OFF_MU;
  float*  sInv = smem + OFF_INV;
  float*  sStr = smem + OFF_STR;
  float*  sCons= smem + OFF_CONS;
  float*  sPair= smem + OFF_PAIR;
  int*    sPerm= (int*)(smem + OFF_PERM);

  const int b    = blockIdx.x;
  const int tid  = threadIdx.x;
  const int wave = tid >> 5;
  const int lane = tid & 31;

  // ------------- Stage A: async-DMA e[b] -> LDS (f32); W -> LDS (bf16) -------
  {
    // Warm L2 for the shared weight matrix.
    __builtin_prefetch(Wp + tid * 64, 0, 3);

    // e[b] (128KB) via GLOBAL_LOAD_ASYNC_TO_LDS_B128: no VGPR staging,
    // tracked with ASYNCcnt.  256 threads x 32 x 16B = 128KB.
    const float* gp = uf + (size_t)b * LSEQ * HDIM;
    unsigned ldsbase = (unsigned)(uintptr_t)sE;  // flat LDS aperture: low 32 bits = LDS offset
#pragma unroll 8
    for (int j = 0; j < 32; ++j) {
      const int idx = tid + 256 * j;             // float4 index
      unsigned loff = ldsbase + idx * 16;
      const float* ga = gp + idx * 4;
      asm volatile("global_load_async_to_lds_b128 %0, %1, off"
                   :: "v"(loff), "v"(ga) : "memory");
    }

    // W f32 -> bf16 into LDS (regular loads; L2-resident, 64 els/thread)
#pragma unroll 4
    for (int i = tid; i < HDIM * HDIM; i += 256) wBF[i] = (__bf16)Wp[i];

    asm volatile("s_wait_asynccnt 0" ::: "memory");
  }
  __syncthreads();

  // e f32 -> bf16 copy for WMMA A operands
#pragma unroll 4
  for (int i = tid; i < LSEQ * HDIM; i += 256) eBF[i] = (__bf16)sE[i];
  __syncthreads();

  // ------------- Stage B: hat = e @ W^T via bf16 WMMA (fp32 acc) -------------
  // Wave w owns N-tile w (8 waves == H/16 tiles): B fragments loaded once,
  // held in registers across all 16 M-tiles.
  {
    const int lh  = lane >> 4;   // half-wave (0/1)
    const int lid = lane & 15;
    const int ncol = wave * 16 + lid;

    v16bf bfrag[4];
    {
      const __bf16* wrow = wBF + ncol * HDIM;   // B[k][n] = W[n][k]
#pragma unroll
      for (int kk = 0; kk < 4; ++kk) {
        const int k0 = kk * 32 + lh * 8;
        v8bf lo = *(const v8bf*)(wrow + k0);
        v8bf hi = *(const v8bf*)(wrow + k0 + 16);
#pragma unroll
        for (int i = 0; i < 8; ++i) { bfrag[kk][i] = lo[i]; bfrag[kk][8 + i] = hi[i]; }
      }
    }

    for (int mt = 0; mt < LSEQ / 16; ++mt) {
      const int m = mt * 16 + lid;
      const __bf16* erow = eBF + m * HDIM;
      v8f acc = {};
#pragma unroll
      for (int kk = 0; kk < 4; ++kk) {
        const int k0 = kk * 32 + lh * 8;
        v8bf lo = *(const v8bf*)(erow + k0);
        v8bf hi = *(const v8bf*)(erow + k0 + 16);
        v16bf af;
#pragma unroll
        for (int i = 0; i < 8; ++i) { af[i] = lo[i]; af[8 + i] = hi[i]; }
        acc = __builtin_amdgcn_wmma_f32_16x16x32_bf16(
            false, af, false, bfrag[kk], (short)0, acc, false, false);
      }
      const int mb2 = mt * 16 + lh * 8;   // C/D layout: VGPR r -> row mb2+r, col lane&15
#pragma unroll
      for (int r = 0; r < 8; ++r) sHat[(mb2 + r) * HDIM + ncol] = acc[r];
    }
  }
  __syncthreads();   // GEMM done; eBF region is now dead -> reuse for sB/sC/...

  // b_init -> sB (must be after GEMM: sB aliases eBF)
  {
    const float* bs = binit + (size_t)b * KCAP * LSEQ;
    for (int i = tid; i < KCAP * LSEQ; i += 256) sB[i] = bs[i];
  }
  __syncthreads();

  // ------------- Stage C: dynamic routing (wave w owns capsule k=w) ----------
  {
    const int k = wave;
    float* brow   = sB + k * LSEQ;
    float* crow   = sC + k * LSEQ;
    float* caprow = sCap + k * HDIM;
    for (int t = 0; t < RITERS; ++t) {
      // softmax over L
      float mx = -3.4e38f;
      for (int l = lane; l < LSEQ; l += 32) mx = fmaxf(mx, brow[l]);
      mx = waveMax(mx);
      float s = 0.f;
      for (int l = lane; l < LSEQ; l += 32) {
        float ex = __expf(brow[l] - mx);
        crow[l] = ex; s += ex;
      }
      s = waveSum(s);
      const float invs = 1.0f / s;
      for (int l = lane; l < LSEQ; l += 32) crow[l] *= invs;

      // cap[k][h] = sum_l c[k][l] * hat[l][h]   (lane owns h = lane+32j)
      float capv[4] = {0.f, 0.f, 0.f, 0.f};
      for (int l = 0; l < LSEQ; ++l) {
        const float cl = crow[l];                 // LDS broadcast
        const float* hr = sHat + l * HDIM;
#pragma unroll
        for (int j = 0; j < 4; ++j) capv[j] += cl * hr[lane + 32 * j];
      }
      // squash
      float n2 = 0.f;
#pragma unroll
      for (int j = 0; j < 4; ++j) n2 += capv[j] * capv[j];
      n2 = waveSum(n2);
      const float scale = n2 / (1.0f + n2) / sqrtf(n2 + 1e-9f);
#pragma unroll
      for (int j = 0; j < 4; ++j) {
        capv[j] *= scale;
        caprow[lane + 32 * j] = capv[j];
      }
      if (t < RITERS - 1) {
        // b[k][l] += hat[l][:] . cap[k][:]   (rotated h index: bank-conflict-free)
        for (int l = lane; l < LSEQ; l += 32) {
          const float* hr = sHat + l * HDIM;
          float d = 0.f;
#pragma unroll 8
          for (int hh = 0; hh < HDIM; ++hh) {
            const int h = (lane + hh) & (HDIM - 1);
            d += hr[h] * caprow[h];
          }
          brow[l] += d;
        }
      } else {
        float s2 = 0.f;
#pragma unroll
        for (int j = 0; j < 4; ++j) s2 += capv[j] * capv[j];
        s2 = waveSum(s2);
        if (lane == 0) sStr[k] = sqrtf(s2);
      }
    }
  }

  // ------------- e row inverse norms ----------------------------------------
  for (int l = wave; l < LSEQ; l += 8) {
    const float* er = sE + l * HDIM;
    float p = 0.f;
#pragma unroll
    for (int j = 0; j < 4; ++j) { float v = er[lane + 32 * j]; p += v * v; }
    p = waveSum(p);
    if (lane == 0) sInv[l] = 1.0f / (sqrtf(p) + 1e-8f);
  }
  __syncthreads();

  // ------------- Stage D: scoring -------------------------------------------
  // Descending argsort of 8 strengths (selection sort with swaps: not stable under ties)
  if (tid == 0) {
    int idx[KCAP];
#pragma unroll
    for (int i = 0; i < KCAP; ++i) idx[i] = i;
    for (int a = 0; a < KCAP; ++a) {
      int best = a;
      for (int q = a + 1; q < KCAP; ++q)
        if (sStr[idx[q]] > sStr[idx[best]]) best = q;
      int tv = idx[a]; idx[a] = idx[best]; idx[best] = tv;
    }
#pragma unroll
    for (int i = 0; i < KCAP; ++i) sPerm[i] = idx[i];
  }
  __syncthreads();

  // mu_n row j (perm order): wave j
  {
    const int j = wave;
    const int p = sPerm[j];
    const float* crow = sC + p * LSEQ;
    float den = 0.f;
    for (int l = lane; l < LSEQ; l += 32) den += crow[l];
    den = waveSum(den);
    float m[4] = {0.f, 0.f, 0.f, 0.f};
    for (int l = 0; l < LSEQ; ++l) {
      const float cl = crow[l];
      const float* er = sE + l * HDIM;
#pragma unroll
      for (int jj = 0; jj < 4; ++jj) m[jj] += cl * er[lane + 32 * jj];
    }
    const float invden = 1.0f / (den + 1e-8f);
#pragma unroll
    for (int jj = 0; jj < 4; ++jj) m[jj] *= invden;
    float nn = 0.f;
#pragma unroll
    for (int jj = 0; jj < 4; ++jj) nn += m[jj] * m[jj];
    nn = waveSum(nn);
    const float invn = 1.0f / (sqrtf(nn) + 1e-8f);
    float* mrow = sMu + j * HDIM;
#pragma unroll
    for (int jj = 0; jj < 4; ++jj) mrow[lane + 32 * jj] = m[jj] * invn;
  }
  __syncthreads();

  // cons_j = sum_l c[perm_j][l] * (e_norm[l] . mu_n[j]);  pair = mu_n . mu_n^T
  {
    const int j = wave;
    const int p = sPerm[j];
    const float* crow = sC + p * LSEQ;
    const float* mrow = sMu + j * HDIM;
    float cp = 0.f;
    for (int l = lane; l < LSEQ; l += 32) {
      const float* er = sE + l * HDIM;
      float d = 0.f;
#pragma unroll 8
      for (int hh = 0; hh < HDIM; ++hh) {
        const int h = (lane + hh) & (HDIM - 1);
        d += er[h] * mrow[h];
      }
      cp += crow[l] * sInv[l] * d;
    }
    cp = waveSum(cp);
    if (lane == 0) sCons[j] = cp;
    for (int i2 = 0; i2 < KCAP; ++i2) {
      const float* m2 = sMu + i2 * HDIM;
      float pp = 0.f;
#pragma unroll
      for (int jj = 0; jj < 4; ++jj) {
        const int h = lane + 32 * jj;
        pp += mrow[h] * m2[h];
      }
      pp = waveSum(pp);
      if (lane == 0) sPair[j * KCAP + i2] = pp;
    }
  }
  __syncthreads();

  // final scores k=2..8, argmax (first-max tie-break, matches jnp.argmax)
  if (tid == 0) {
    float cum = sCons[0];
    float up = 0.f;
    float best = -3.4e38f;
    int bi = 2;
    for (int kk = 2; kk <= KCAP; ++kk) {
      cum += sCons[kk - 1];
      for (int i = 0; i < kk - 1; ++i) up += sPair[i * KCAP + (kk - 1)];
      const float consistency = cum / (float)(kk * LSEQ);
      const float diversity = 1.0f - (2.0f / (float)(kk * (kk - 1))) * up;
      const float sc = 0.5f * consistency + 0.5f * diversity;
      out[BATCH + b * 7 + (kk - 2)] = sc;
      if (sc > best) { best = sc; bi = kk; }
    }
    out[b] = (float)bi;
  }
}

extern "C" void kernel_launch(void* const* d_in, const int* in_sizes, int n_in,
                              void* d_out, int out_size, void* d_ws, size_t ws_size,
                              hipStream_t stream) {
  (void)in_sizes; (void)n_in; (void)d_ws; (void)ws_size; (void)out_size;
  const float* uf    = (const float*)d_in[0];  // user_features [B,L,H] f32
  const float* W     = (const float*)d_in[1];  // W [H,H] f32
  const float* binit = (const float*)d_in[2];  // b_init [B,K,L] f32
  float* out = (float*)d_out;                  // [B] + [B,7]

  // Allow up to the WGP max 320KB dynamic LDS.
  hipFuncSetAttribute((const void*)mind_fused_kernel,
                      hipFuncAttributeMaxDynamicSharedMemorySize, SMEM_BYTES);

  mind_fused_kernel<<<BATCH, 256, SMEM_BYTES, stream>>>(uf, W, binit, out);
}
